// SetConvEncoder_10806137717277
// MI455X (gfx1250) — matrix operand order repaired and checked by reading, not compiled, not run on hardware
//
#include <hip/hip_runtime.h>

// CDNA5 WMMA fragment types (wave32)
typedef __attribute__((ext_vector_type(16))) _Float16 v16h;
typedef __attribute__((ext_vector_type(8)))  _Float16 v8h;
typedef __attribute__((ext_vector_type(8)))  float    v8f;

// Static problem config (matches reference setup_inputs / _make_grid)
#define M_BATCH 2
#define NC      1024          // context points (K dimension)
#define DY      16            // data channels (ones channel handled separately)
#define GRID_N  256
#define NG      (GRID_N * GRID_N)   // 65536 grid points
#define PPU_INV (1.0f / 64.0f)
#define HALF_N  128
#define X_MID   1.0f

#define WAVES_PER_BLOCK 8
#define GPTS_PER_BLOCK  (WAVES_PER_BLOCK * 16)   // 128 grid points per block
#define YLDS_STRIDE     (NC + 8)                 // pad 16B -> 4-bank skew per channel row

__global__ __launch_bounds__(256) void
setconv_encoder_wmma(const float* __restrict__ xc,
                     const float* __restrict__ yc,
                     const float* __restrict__ lsp,
                     float* __restrict__ out)
{
    // LDS: normalized context coords (ax, ay, |a|^2, pad) + transposed f16 yc
    __shared__ float4 xcs[NC];                                     // 16 KB
    __shared__ __align__(16) _Float16 ylds[DY][YLDS_STRIDE];       // ~33 KB

    const int tid   = threadIdx.x;
    const int nblkg = NG / GPTS_PER_BLOCK;               // 512 blocks per batch
    const int mb    = blockIdx.x / nblkg;                // batch index
    const int gblk  = blockIdx.x % nblkg;
    const int gbase_block = gblk * GPTS_PER_BLOCK;

    // lengthscale = 1e-5 + softplus(param), per dim
    const float il0 = 1.0f / (1e-5f + log1pf(__expf(lsp[0])));
    const float il1 = 1.0f / (1e-5f + log1pf(__expf(lsp[1])));

    // Stage normalized xc into LDS
    for (int k = tid; k < NC; k += 256) {
        float ax = xc[((size_t)mb * NC + k) * 2 + 0] * il0;
        float ay = xc[((size_t)mb * NC + k) * 2 + 1] * il1;
        xcs[k] = make_float4(ax, ay, ax * ax + ay * ay, 0.0f);
    }
    // Stage yc transposed [ch][k] as f16 (coalesced global reads)
    for (int idx = tid; idx < NC * DY; idx += 256) {
        int k = idx >> 4, ch = idx & 15;
        ylds[ch][k] = (_Float16)yc[((size_t)mb * NC + k) * DY + ch];
    }
    __syncthreads();

    // ---- output region 0: x_grid_b [m,256,256,2] (coordinates) ----
    {
        int p = tid >> 1, comp = tid & 1;
        int g = gbase_block + p;
        int rc = comp ? (g & (GRID_N - 1)) : (g >> 8);
        out[((size_t)mb * NG + g) * 2 + comp] =
            X_MID + (float)(rc - HALF_N) * PPU_INV;
    }

    float* out_z = out + (size_t)M_BATCH * NG * 2;   // z_grid region [m,256,256,17]

    const int wave = tid >> 5;
    const int lane = tid & 31;
    const int hb   = lane >> 4;      // lane half (0: lanes 0-15, 1: lanes 16-31)
    const int mrow = lane & 15;      // row (for A/density) / column N (for B/C)

    const int gbase = gbase_block + wave * 16;
    const int g     = gbase + mrow;

    // This lane's grid point, normalized by lengthscale
    const float gx  = (X_MID + (float)((g >> 8) - HALF_N) * PPU_INV) * il0;
    const float gy  = (X_MID + (float)((g & (GRID_N - 1)) - HALF_N) * PPU_INV) * il1;
    const float gsq = gx * gx + gy * gy;
    const float ngx = -2.0f * gx, ngy = -2.0f * gy;

    v8f   acc  = {};     // 16x16 f32 accumulator tile
    float dsum = 0.0f;   // density channel partial (this lane's K subset)

    for (int ks = 0; ks < NC; ks += 32) {
        // ---- A fragment: 16x32 f16 weights, CDNA5 layout ----
        // a[j]   -> K = ks + hb*8 + j        (j = 0..7)
        // a[8+j] -> K = ks + 16 + hb*8 + j
        v16h a;
        const int kA = ks + hb * 8;
        #pragma unroll
        for (int j = 0; j < 8; ++j) {
            float4 c0 = xcs[kA + j];
            float d2a = gsq + c0.z + ngx * c0.x + ngy * c0.y;
            float wa  = __expf(-0.5f * d2a);
            dsum += wa;
            a[j] = (_Float16)wa;

            float4 c1 = xcs[kA + 16 + j];
            float d2b = gsq + c1.z + ngx * c1.x + ngy * c1.y;
            float wb  = __expf(-0.5f * d2b);
            dsum += wb;
            a[8 + j] = (_Float16)wb;
        }

        // ---- B fragment: 32x16 f16, lane holds 16 K at N=mrow ----
        // b[e] -> (K = ks + hb*8 + e, N = mrow); b[8+e] -> K = ks + 16 + hb*8 + e
        const v8h* bp = (const v8h*)&ylds[mrow][ks + hb * 8];
        v8h b0 = bp[0], b1 = bp[2];
        v16h b = __builtin_shufflevector(b0, b1, 0, 1, 2, 3, 4, 5, 6, 7,
                                                 8, 9, 10, 11, 12, 13, 14, 15);

        acc = __builtin_amdgcn_wmma_f32_16x16x32_f16(
            /*neg_a=*/false, a, /*neg_b=*/false, b,
            /*c_mod=*/(short)0, acc, /*reuse_a=*/false, /*reuse_b=*/false);
    }

    // density channel: combine complementary K subsets across lane halves
    dsum += __shfl_xor(dsum, 16, 32);

    // ---- store z: acc[r] = z[gbase + r + 8*hb][mrow] ----
    #pragma unroll
    for (int r = 0; r < 8; ++r) {
        int row = r + 8 * hb;
        out_z[((size_t)mb * NG + gbase + row) * 17 + mrow] = acc[r];
    }
    if (hb == 0) {
        out_z[((size_t)mb * NG + gbase + mrow) * 17 + 16] = dsum;
    }
}

extern "C" void kernel_launch(void* const* d_in, const int* in_sizes, int n_in,
                              void* d_out, int out_size, void* d_ws, size_t ws_size,
                              hipStream_t stream)
{
    (void)in_sizes; (void)n_in; (void)out_size; (void)d_ws; (void)ws_size;
    const float* xc  = (const float*)d_in[0];
    const float* yc  = (const float*)d_in[1];
    // d_in[2] (xt) is unused by the reference output
    const float* lsp = (const float*)d_in[3];
    float* out = (float*)d_out;

    dim3 grid(M_BATCH * (NG / GPTS_PER_BLOCK));   // 1024 blocks
    dim3 block(256);                              // 8 wave32 waves
    hipLaunchKernelGGL(setconv_encoder_wmma, grid, block, 0, stream,
                       xc, yc, lsp, out);
}
